// SelfAttention_55001351193149
// MI455X (gfx1250) — matrix operand order repaired and checked by reading, not compiled, not run on hardware
//
#include <hip/hip_runtime.h>
#include <hip/hip_bf16.h>

// ---------------------------------------------------------------------------
// Types for CDNA5 WMMA (wave32): D(16x16 f32) = A(16x32 bf16) * B(32x16 bf16) + C
// ---------------------------------------------------------------------------
typedef __bf16 bf16_t;
typedef __attribute__((ext_vector_type(16))) __bf16 v16bf;
typedef __attribute__((ext_vector_type(8)))  __bf16 v8bf;
typedef __attribute__((ext_vector_type(8)))  float  v8f;
typedef __attribute__((ext_vector_type(4))) unsigned int u32x4;
typedef __attribute__((ext_vector_type(8))) int          i32x8;
typedef __attribute__((ext_vector_type(4))) int          i32x4;

#define SEQ 2048
#define HID 1024
#define NHEAD 16
#define HDIM 64

__device__ __forceinline__ v8f wmma_bf16(v16bf a, v16bf b, v8f c) {
  return __builtin_amdgcn_wmma_f32_16x16x32_bf16(false, a, false, b, (short)0, c,
                                                 false, false);
}

// A-matrix fragment (16x32, M=row per lane&15). ISA layout:
//   lanes 0-15: elems 0-7 -> K=0..7,  elems 8-15 -> K=16..23
//   lanes16-31: elems 0-7 -> K=8..15, elems 8-15 -> K=24..31
__device__ __forceinline__ v16bf ld_fragA(const bf16_t* row, int hb) {
  v8bf lo = *(const v8bf*)(row + hb * 8);
  v8bf hi = *(const v8bf*)(row + 16 + hb * 8);
  v16bf r;
#pragma unroll
  for (int i = 0; i < 8; ++i) { r[i] = lo[i]; r[i + 8] = hi[i]; }
  return r;
}

// B-matrix fragment (32x16, N=col per lane&15). Same K order as A:
//   lanes 0-15: elems 0-7 -> K=0..7,  elems 8-15 -> K=16..23
//   lanes16-31: elems 0-7 -> K=8..15, elems 8-15 -> K=24..31
__device__ __forceinline__ v16bf ld_fragB(const bf16_t* col, int hb) {
  v8bf lo = *(const v8bf*)(col + hb * 8);
  v8bf hi = *(const v8bf*)(col + 16 + hb * 8);
  v16bf r;
#pragma unroll
  for (int i = 0; i < 8; ++i) { r[i] = lo[i]; r[i + 8] = hi[i]; }
  return r;
}

// Generic shared-space pointer -> 32-bit LDS byte address (aperture low bits).
__device__ __forceinline__ unsigned int lds_addr32(const void* p) {
  return (unsigned int)(uintptr_t)p;
}

// ---------------------------------------------------------------------------
// Generic GEMM:  C[M,N] = scale * (A[M,K] @ W[N,K]^T + bias[N])
// Block tile 128x64, 8 waves, each wave 32x32 (2x2 WMMA subtiles), BK=32.
// ---------------------------------------------------------------------------
#define TILE_M 128
#define TILE_N 64
#define TILE_K 32
#define LDT 40  // padded LDS stride (elems) for K=32 tiles; 80B, 16B-aligned

template <bool A_BF16, bool OUT_BF16>
__global__ __launch_bounds__(256) void gemm_bias_kernel(
    const void* __restrict__ Aptr, const float* __restrict__ W,
    const float* __restrict__ bias, void* __restrict__ Cptr,
    int M, int N, int K, float scale) {
  __shared__ __align__(16) bf16_t As[TILE_M * LDT];
  __shared__ __align__(16) bf16_t Ws[TILE_N * LDT];

  const int tid  = threadIdx.x;
  const int lane = tid & 31;
  const int wid  = tid >> 5;
  const int l16  = lane & 15;
  const int hb   = lane >> 4;
  const int wm   = wid & 3;   // wave row group: 4 x 32 rows
  const int wn   = wid >> 2;  // wave col group: 2 x 32 cols
  const int bm   = blockIdx.y * TILE_M;
  const int bn   = blockIdx.x * TILE_N;

  v8f acc[2][2] = {};

  for (int k0 = 0; k0 < K; k0 += TILE_K) {
    const bool more = (k0 + TILE_K) < K;
    // ---- stage A tile (convert to bf16 if needed) ----
    if (A_BF16) {
      const bf16_t* A = (const bf16_t*)Aptr;
#pragma unroll
      for (int i = tid; i < TILE_M * (TILE_K / 8); i += 256) {
        int r = i >> 2;            // 4 groups of 8 per row
        int c = (i & 3) * 8;
        const bf16_t* src = A + (size_t)(bm + r) * K + k0 + c;
        *(v8bf*)(&As[r * LDT + c]) = *(const v8bf*)src;
        if (more && (i & 3) == 0) __builtin_prefetch(src + TILE_K, 0, 1);
      }
    } else {
      const float* A = (const float*)Aptr;
#pragma unroll
      for (int i = tid; i < TILE_M * (TILE_K / 4); i += 256) {
        int r = i >> 3;            // 8 float4 per row
        int c = (i & 7) * 4;
        const float* src = A + (size_t)(bm + r) * K + k0 + c;
        float4 v = *(const float4*)src;
        bf16_t* d = &As[r * LDT + c];
        d[0] = (bf16_t)v.x; d[1] = (bf16_t)v.y;
        d[2] = (bf16_t)v.z; d[3] = (bf16_t)v.w;
        if (more && (i & 7) == 0) __builtin_prefetch(src + TILE_K, 0, 1);
      }
    }
    // ---- stage W tile (fp32 -> bf16) ----
#pragma unroll
    for (int i = tid; i < TILE_N * (TILE_K / 4); i += 256) {
      int r = i >> 3;
      int c = (i & 7) * 4;
      const float* src = W + (size_t)(bn + r) * K + k0 + c;
      float4 v = *(const float4*)src;
      bf16_t* d = &Ws[r * LDT + c];
      d[0] = (bf16_t)v.x; d[1] = (bf16_t)v.y;
      d[2] = (bf16_t)v.z; d[3] = (bf16_t)v.w;
      if (more && (i & 7) == 0) __builtin_prefetch(src + TILE_K, 0, 1);
    }
    __syncthreads();

    v16bf a0 = ld_fragA(&As[(wm * 32 + l16) * LDT], hb);
    v16bf a1 = ld_fragA(&As[(wm * 32 + 16 + l16) * LDT], hb);
    v16bf b0 = ld_fragB(&Ws[(wn * 32 + l16) * LDT], hb);
    v16bf b1 = ld_fragB(&Ws[(wn * 32 + 16 + l16) * LDT], hb);

    acc[0][0] = wmma_bf16(a0, b0, acc[0][0]);
    acc[0][1] = wmma_bf16(a0, b1, acc[0][1]);
    acc[1][0] = wmma_bf16(a1, b0, acc[1][0]);
    acc[1][1] = wmma_bf16(a1, b1, acc[1][1]);

    __syncthreads();
  }

  // ---- epilogue: C layout row = v + 8*hb, col = l16 ----
#pragma unroll
  for (int sm = 0; sm < 2; ++sm) {
#pragma unroll
    for (int sn = 0; sn < 2; ++sn) {
      int col = bn + wn * 32 + sn * 16 + l16;
      float bc = bias[col];
      int row0 = bm + wm * 32 + sm * 16 + 8 * hb;
#pragma unroll
      for (int v = 0; v < 8; ++v) {
        float val = (acc[sm][sn][v] + bc) * scale;
        size_t idx = (size_t)(row0 + v) * N + col;
        if (OUT_BF16) ((bf16_t*)Cptr)[idx] = (bf16_t)val;
        else          ((float*)Cptr)[idx]  = val;
      }
    }
  }
}

// ---------------------------------------------------------------------------
// Flash attention (causal), one head x 128 query rows per block, 8 waves.
// Q is pre-scaled by 1/sqrt(HD). Q/K/V/Ctx are bf16 [SEQ][HID] (head = 64 cols).
// K chunk is staged with the Tensor Data Mover (ISA ch.8 D# descriptor):
//   2D tile 64x32 elems (2B), tensor 1024x2048, row stride 1024 elems,
//   LDS padding 4 dwords (8 bf16) after every 32 dwords -> Ks stride 72.
// ---------------------------------------------------------------------------
__global__ __launch_bounds__(256) void attn_kernel(
    const bf16_t* __restrict__ Q, const bf16_t* __restrict__ K,
    const bf16_t* __restrict__ V, bf16_t* __restrict__ Ctx) {
  __shared__ __align__(16) bf16_t Ks[32 * 72];      // K chunk, row-major [kv][hd]
  __shared__ __align__(16) bf16_t Vt[64 * 40];      // V chunk transposed [hd][kv]
  __shared__ __align__(16) bf16_t Ps[8][16 * 40];   // per-wave P scratch

  const int tid  = threadIdx.x;
  const int lane = tid & 31;
  const int wid  = tid >> 5;
  const int l16  = lane & 15;
  const int hb   = lane >> 4;
  const int h    = blockIdx.y;
  const int q0b  = blockIdx.x * 128;
  const int q0w  = q0b + wid * 16;   // this wave's first query row

  // Q fragments: rows q0w+0..15, HD split into two K=32 chunks
  const bf16_t* qrow = Q + (size_t)(q0w + l16) * HID + h * HDIM;
  v16bf qf0 = ld_fragA(qrow, hb);
  v16bf qf1 = ld_fragA(qrow + 32, hb);

  v8f acc[4] = {};          // ctx 16 x 64 (4 tiles of 16x16)
  float m_i[8], l_i[8];
#pragma unroll
  for (int v = 0; v < 8; ++v) { m_i[v] = -1e30f; l_i[v] = 0.f; }

  const int kv_end = q0b + 128;
  for (int kv0 = 0; kv0 < kv_end; kv0 += 32) {
    // ---- TDM: K chunk [kv0..kv0+31][h*64..h*64+63] -> Ks (padded rows) ----
    if (wid == 0) {
      unsigned long long ga =
          (unsigned long long)(uintptr_t)(K + (size_t)kv0 * HID + h * HDIM);
      u32x4 g0 = {1u,                             // count=1, no gather
                  lds_addr32(&Ks[0]),             // lds_addr (bytes)
                  (unsigned int)ga,               // global_addr[31:0]
                  (unsigned int)((ga >> 32) & 0x1FFFFFFu) | 0x80000000u}; // type=2
      i32x8 g1 = {0x07110000,   // data_size=2B, pad_en, pad_interval=32dw, pad=4dw
                  0x04000000,   // tensor_dim0 = 1024 (bits 63:48)
                  0x08000000,   // tensor_dim1 = 2048 (bits 95:80)
                  0x00400000,   // tile_dim0 = 64 (bits 127:112)
                  32,           // tile_dim1 = 32 (bits 143:128)
                  1024,         // tensor_dim0_stride[31:0]
                  0, 0};
      i32x4 gz4 = {0, 0, 0, 0};
      i32x8 gz8 = {0, 0, 0, 0, 0, 0, 0, 0};
      __builtin_amdgcn_tensor_load_to_lds(g0, g1, gz4, gz4, gz8, 0);
    }
    // ---- stage V chunk (transposed) cooperatively ----
    {
      int r = tid >> 3;           // kv row 0..31
      int c = (tid & 7) * 8;      // hd col group
      const size_t src = (size_t)(kv0 + r) * HID + h * HDIM + c;
      v8bf vv = *(const v8bf*)(V + src);
#pragma unroll
      for (int j = 0; j < 8; ++j) Vt[(c + j) * 40 + r] = vv[j];
    }
    if (wid == 0) __builtin_amdgcn_s_wait_tensorcnt(0);
    __syncthreads();

    if (kv0 <= q0w + 15) {   // wave-uniform: chunk not fully masked
      // ---- scores: two 16x16 tiles over 32 kv cols ----
      v8f s0 = {}, s1 = {};
      {
        v16bf kb;
        kb = ld_fragB(&Ks[l16 * 72], hb);            s0 = wmma_bf16(qf0, kb, s0);
        kb = ld_fragB(&Ks[l16 * 72 + 32], hb);       s0 = wmma_bf16(qf1, kb, s0);
        kb = ld_fragB(&Ks[(16 + l16) * 72], hb);     s1 = wmma_bf16(qf0, kb, s1);
        kb = ld_fragB(&Ks[(16 + l16) * 72 + 32], hb);s1 = wmma_bf16(qf1, kb, s1);
      }
      // ---- causal mask + online softmax (row = q0w + v + 8*hb, col via l16) ----
      const int rbase = q0w + 8 * hb;
#pragma unroll
      for (int v = 0; v < 8; ++v) {
        const int r = rbase + v;
        float a0 = ((kv0 + l16)      <= r) ? s0[v] : -1e30f;
        float a1 = ((kv0 + 16 + l16) <= r) ? s1[v] : -1e30f;
        float mv = fmaxf(a0, a1);
        mv = fmaxf(mv, __shfl_xor(mv, 1, 32));
        mv = fmaxf(mv, __shfl_xor(mv, 2, 32));
        mv = fmaxf(mv, __shfl_xor(mv, 4, 32));
        mv = fmaxf(mv, __shfl_xor(mv, 8, 32));
        float mnew  = fmaxf(m_i[v], mv);
        float alpha = __expf(m_i[v] - mnew);
        float e0    = __expf(a0 - mnew);
        float e1    = __expf(a1 - mnew);
        float ls    = e0 + e1;
        ls += __shfl_xor(ls, 1, 32);
        ls += __shfl_xor(ls, 2, 32);
        ls += __shfl_xor(ls, 4, 32);
        ls += __shfl_xor(ls, 8, 32);
        l_i[v] = l_i[v] * alpha + ls;
        m_i[v] = mnew;
#pragma unroll
        for (int t = 0; t < 4; ++t) acc[t][v] *= alpha;
        // park P (bf16) in per-wave LDS in A-matrix memory order [16][32]
        Ps[wid][(v + 8 * hb) * 40 + l16]      = (bf16_t)e0;
        Ps[wid][(v + 8 * hb) * 40 + 16 + l16] = (bf16_t)e1;
      }
      // ---- PV: ctx += P(16x32) x V(32x64) ----
      v16bf pf = ld_fragA(&Ps[wid][l16 * 40], hb);
#pragma unroll
      for (int t = 0; t < 4; ++t) {
        v16bf vf = ld_fragB(&Vt[(t * 16 + l16) * 40], hb);
        acc[t] = wmma_bf16(pf, vf, acc[t]);
      }
    }
    __syncthreads();
  }

  // ---- finalize: ctx /= l, store bf16 ----
#pragma unroll
  for (int t = 0; t < 4; ++t) {
    int col = h * HDIM + t * 16 + l16;
#pragma unroll
    for (int v = 0; v < 8; ++v) {
      int row = q0w + v + 8 * hb;
      Ctx[(size_t)row * HID + col] = (bf16_t)(acc[t][v] / l_i[v]);
    }
  }
}

// ---------------------------------------------------------------------------
// Launcher. Inputs (dict order): hidden_states, Wq, bq, Wk, bk, Wv, bv, Wo, bo,
// attention_mask (ignored: causal is known). Output: [S,1,H] fp32.
// Workspace: Q,K,V,Ctx bf16 [SEQ][HID] -> 4 * 4MB = 16MB.
// ---------------------------------------------------------------------------
extern "C" void kernel_launch(void* const* d_in, const int* in_sizes, int n_in,
                              void* d_out, int out_size, void* d_ws, size_t ws_size,
                              hipStream_t stream) {
  (void)in_sizes; (void)n_in; (void)out_size; (void)ws_size;
  const float* hs = (const float*)d_in[0];
  const float* Wq = (const float*)d_in[1];
  const float* bq = (const float*)d_in[2];
  const float* Wk = (const float*)d_in[3];
  const float* bk = (const float*)d_in[4];
  const float* Wv = (const float*)d_in[5];
  const float* bv = (const float*)d_in[6];
  const float* Wo = (const float*)d_in[7];
  const float* bo = (const float*)d_in[8];
  float* out = (float*)d_out;

  bf16_t* Qb = (bf16_t*)d_ws;
  bf16_t* Kb = Qb + (size_t)SEQ * HID;
  bf16_t* Vb = Kb + (size_t)SEQ * HID;
  bf16_t* Cb = Vb + (size_t)SEQ * HID;

  dim3 blk(256);
  dim3 gg(HID / TILE_N, SEQ / TILE_M);  // (16, 16)

  gemm_bias_kernel<false, true><<<gg, blk, 0, stream>>>(hs, Wq, bq, Qb, SEQ, HID, HID, 0.125f);
  gemm_bias_kernel<false, true><<<gg, blk, 0, stream>>>(hs, Wk, bk, Kb, SEQ, HID, HID, 1.0f);
  gemm_bias_kernel<false, true><<<gg, blk, 0, stream>>>(hs, Wv, bv, Vb, SEQ, HID, HID, 1.0f);

  attn_kernel<<<dim3(SEQ / 128, NHEAD), blk, 0, stream>>>(Qb, Kb, Vb, Cb);

  gemm_bias_kernel<true, false><<<gg, blk, 0, stream>>>(Cb, Wo, bo, out, SEQ, HID, HID, 1.0f);
}
